// Codec_15736760172741
// MI455X (gfx1250) — matrix operand order repaired and checked by reading, not compiled, not run on hardware
//
#include <hip/hip_runtime.h>
#include <hip/hip_bf16.h>

typedef __attribute__((ext_vector_type(16))) _Float16 v16h;
typedef __attribute__((ext_vector_type(8)))  float    v8f;
typedef __attribute__((ext_vector_type(4)))  unsigned v4u;
typedef __attribute__((ext_vector_type(8)))  int      v8i;
typedef __attribute__((ext_vector_type(4)))  int      v4i;

#define IMG_H 512
#define IMG_W 512
#define NIMG  24
#define HW    (IMG_H * IMG_W)
#define ROWS_PER_BLOCK 8
#define ROWBUF_ROWS 11
#define ROWW  (IMG_W + 6)                         /* 3-col zero halo each side */
#define ROWBUF_BYTES 22848                        /* 11*518*4 = 22792, 64B aligned */
#define WAVE_LDS 3584                             /* actN 1024 + actP 2048 + epi 512 */
#define LDS_TOTAL (ROWBUF_BYTES + 8 * WAVE_LDS)   /* 51520 bytes */

struct CParams {
  const float *w00T, *b00T, *w00L, *w01, *b01, *w02, *b02,
              *w03, *b03, *w04, *b04, *w05, *b05;
};

__device__ __forceinline__ void ldsFence() {
  asm volatile("s_wait_dscnt 0" ::: "memory");
}

// ---- WMMA fragment helpers (C/D layout per CDNA5 ISA §7.12.2; A/B use this code's contiguous k order) ----

// 16-bit A matrix 16x32 in this code's contiguous k order: lane holds row M=lane%16.
// lanes 0-15: K=e (0..15)
// lanes16-31: K=e+16 (16..31)
template <typename F>
__device__ __forceinline__ v16h makeA(F w, int lane) {
  v16h a{};
  const int M = lane & 15, hi = lane >> 4;
#pragma unroll
  for (int e = 0; e < 16; ++e) {
    const int k = e + 16 * hi;
    a[e] = (_Float16)w(M, k);
  }
  return a;
}

// C/D f32 16x16: vgpr v of lane L holds (M = v + 8*(L>>4), N = L&15)
template <typename F>
__device__ __forceinline__ v8f makeC(F b, int lane) {
  v8f c{};
  const int hi = lane >> 4;
#pragma unroll
  for (int v = 0; v < 8; ++v) c[v] = b(v + 8 * hi);
  return c;
}

__device__ __forceinline__ v8f wmma16(v16h a, v16h b, v8f c) {
  return __builtin_amdgcn_wmma_f32_16x16x32_f16(false, a, false, b, (short)0, c,
                                                false, false);
}

__device__ __forceinline__ void leaky8(v8f &d) {
#pragma unroll
  for (int v = 0; v < 8; ++v) d[v] = d[v] > 0.f ? d[v] : 0.01f * d[v];
}

// Store a D fragment into LDS staging actN[pixel N][k] (f16), K rows 0..15.
__device__ __forceinline__ void storeToB(_Float16 *actN, const v8f &d, int lane) {
  const int N = lane & 15, hi = lane >> 4;
#pragma unroll
  for (int v = 0; v < 8; ++v) actN[N * 32 + v + 8 * hi] = (_Float16)d[v];
}

// 16-bit B matrix 32x16: lane holds col N=lane%16; lanes 0-15: K=0..15,
// lanes 16-31: K=16..31, contiguous -> one 32B LDS read.
__device__ __forceinline__ v16h loadB(const _Float16 *actN, int lane) {
  const int N = lane & 15, hi = lane >> 4;
  return *(const v16h *)(actN + N * 32 + hi * 16);
}

// -------------------------------------------------------------------------

__global__ void __launch_bounds__(256)
codec_pass(const float *__restrict__ src, float *__restrict__ dst, CParams P) {
  extern __shared__ __align__(64) char smem[];
  const int tid = threadIdx.x;
  const int lane = tid & 31;
  const int wave = tid >> 5;
  const int img = blockIdx.y;
  const int row0 = blockIdx.x * ROWS_PER_BLOCK;
  const int row = row0 + wave;
  const int hi = lane >> 4;
  const int N = lane & 15;

  float *rowbuf = (float *)smem;  // [11][518] f32, halo cols 0-2 / 515-517
  _Float16 *actN = (_Float16 *)(smem + ROWBUF_BYTES + wave * WAVE_LDS);
  float *actP = (float *)(smem + ROWBUF_BYTES + wave * WAVE_LDS + 1024);
  float *epi = (float *)(smem + ROWBUF_BYTES + wave * WAVE_LDS + 3072);

  // zero-fill rows above image top (only first row-block)
  const int firstRL = (row0 == 0) ? 3 : 0;
  for (int i = tid; i < firstRL * ROWW; i += 256) rowbuf[i] = 0.0f;
  // zero-fill left/right halo columns of all loaded rows
  for (int i = tid; i < ROWBUF_ROWS * 6; i += 256) {
    const int r = i / 6, c = i % 6;
    rowbuf[r * ROWW + (c < 3 ? c : (ROWW - 6) + c)] = 0.0f;
  }

  // TDM iterate mode: DMA one 512-elem row per iteration, LDS stride 518 elems
  if (wave == 0) {
    const int rStart = row0 - 3 + firstRL;
    const int nrows = ROWBUF_ROWS - firstRL;
    const unsigned long long ga =
        (unsigned long long)(src + (size_t)img * HW + (size_t)rStart * IMG_W);
    v4u g0{};
    v8i g1{};
    v4i g2{};
    v4i gz4 = {0, 0, 0, 0};
    v8i gz8 = {0, 0, 0, 0, 0, 0, 0, 0};
    g0[0] = 1u;                                        // count=1, normal mode
    g0[1] = (unsigned)((firstRL * ROWW + 3) * 4);      // lds_addr (bytes)
    g0[2] = (unsigned)ga;                              // global_addr lo
    g0[3] = (unsigned)((ga >> 32) & 0x01FFFFFFu) | (2u << 30);  // addr hi | type=2
    g1[0] = (int)((2u << 16) | (1u << 19));            // data_size=4B, iterate_en
    g1[1] = (int)((unsigned)(IMG_W & 0xFFFF) << 16);   // tensor_dim0 lo16
    g1[2] = (int)(((unsigned)IMG_W >> 16) | (1u << 16)); // dim0 hi | dim1 lo = 1
    g1[3] = (int)((unsigned)IMG_W << 16);              // dim1 hi | tile_dim0=512
    g1[4] = 1;                                         // tile_dim1=1, tile_dim2=0
    g1[5] = (int)(unsigned)IMG_W;                      // tensor_dim0_stride
    g1[6] = 0;
    g1[7] = 0;
    g2[0] = 0;                                         // tensor_dim2
    g2[1] = (int)(unsigned)ROWW;                       // lds_addr_increment (elems)
    g2[2] = (int)(unsigned)IMG_W;                      // global_addr_increment lo
    g2[3] = (int)((unsigned)(nrows - 1) << 16);        // iterate_count (0 = 1x)
    __builtin_amdgcn_tensor_load_to_lds(g0, g1, g2, gz4, gz8, 0);
    __builtin_amdgcn_s_wait_tensorcnt(0);
  }
  __syncthreads();

  // ---- A fragments + bias fragments (held in VGPRs for all 32 tiles) ----
  v16h aSp = makeA(
      [&](int m, int k) -> float {
        if (k < 21) return P.w00T[m * 21 + k];            // top taps dy=k/7 dx=k%7
        if (k < 24) return P.w00L[m * 3 + (k - 21)];      // left taps
        return 0.f;
      },
      lane);
  v16h a01 = makeA([&](int m, int k) { return k < 16 ? P.w01[m * 16 + k] : 0.f; }, lane);
  v16h a02 = makeA([&](int m, int k) { return k < 16 ? P.w02[m * 16 + k] : 0.f; }, lane);
  v16h a3l = makeA([&](int m, int k) { return k < 16 ? P.w03[m * 16 + k] : 0.f; }, lane);
  v16h a3h = makeA([&](int m, int k) { return k < 16 ? P.w03[(m + 16) * 16 + k] : 0.f; }, lane);
  v16h a04 = makeA([&](int m, int k) { return k < 4 ? P.w04[m * 4 + k] : 0.f; }, lane);
  v16h a05 = makeA([&](int m, int k) { return (m < 4 && k < 16) ? P.w05[m * 16 + k] : 0.f; }, lane);

  v8f cSp = makeC([&](int m) { return P.b00T[m]; }, lane);
  v8f c01 = makeC([&](int m) { return P.b01[m]; }, lane);
  v8f c02 = makeC([&](int m) { return P.b02[m]; }, lane);
  v8f c3l = makeC([&](int m) { return P.b03[m]; }, lane);
  v8f c3h = makeC([&](int m) { return P.b03[m + 16]; }, lane);
  v8f c04 = makeC([&](int m) { return P.b04[m]; }, lane);
  v8f c05 = makeC([&](int m) { return m < 4 ? P.b05[m] : 0.f; }, lane);

  // zero the K=16..31 half of the B staging buffer (never written by D stores)
  for (int i = lane; i < 256; i += 32) {
    const int n = i >> 4, k = 16 + (i & 15);
    actN[n * 32 + k] = (_Float16)0.f;
  }
  ldsFence();

  // ---- per-wave: 32 tiles of 16 pixels along this row ----
  for (int wt = 0; wt < IMG_W; wt += 16) {
    // B im2col fragment for the spatial (causal) conv; K taps:
    // k=0..20: rows h-3..h-1, cols w-3..w+3 ; k=21..23: row h, cols w-3..w-1
    // Halo layout makes every tap an unconditional LDS load (no bounds checks).
    v16h b{};
    const int base = wt + N;  // padded col index of w-3
#pragma unroll
    for (int e = 0; e < 16; ++e) {
      const int k = e + 16 * hi;
      const int dy = (k < 21) ? (k / 7) : 3;
      const int dx = (k < 21) ? (k % 7) : (k - 21);
      const bool valid = (k < 24);
      const float v = rowbuf[(wave + dy) * ROWW + base + (valid ? dx : 0)];
      b[e] = (_Float16)(valid ? v : 0.f);
    }
    v8f d = wmma16(aSp, b, cSp);
    leaky8(d);
    storeToB(actN, d, lane);
    ldsFence();

    b = loadB(actN, lane);
    d = wmma16(a01, b, c01);
    leaky8(d);
    storeToB(actN, d, lane);
    ldsFence();

    b = loadB(actN, lane);
    d = wmma16(a02, b, c02);
    leaky8(d);
    storeToB(actN, d, lane);
    ldsFence();

    b = loadB(actN, lane);
    v8f d3l = wmma16(a3l, b, c3l);
    v8f d3h = wmma16(a3h, b, c3h);
#pragma unroll
    for (int v = 0; v < 8; ++v) {
      actP[N * 32 + v + 8 * hi] = d3l[v];
      actP[N * 32 + 16 + v + 8 * hi] = d3h[v];
    }
    ldsFence();

    // per-pixel softmax-mix epilogue on lanes 0..15
    if (lane < 16) {
      const int p = lane;
      float ch[32];
#pragma unroll
      for (int c = 0; c < 32; ++c) ch[c] = actP[p * 32 + c];
      float ta2[4];
#pragma unroll
      for (int j = 0; j < 4; ++j) {
        const float *r = &ch[j * 4];
        const float mx = fmaxf(fmaxf(r[0], r[1]), fmaxf(r[2], r[3]));
        const float e0 = __expf(r[0] - mx), e1 = __expf(r[1] - mx);
        const float e2 = __expf(r[2] - mx), e3 = __expf(r[3] - mx);
        const float s = e0 + e1 + e2 + e3;
        const float mix =
            (e0 * ch[16] + e1 * ch[17] + e2 * ch[18] + e3 * ch[19]) / s;
        ta2[j] = mix + ch[20 + j];
      }
      const float mxc = fmaxf(fmaxf(ch[24], ch[25]), fmaxf(ch[26], ch[27]));
      const float f0 = __expf(ch[24] - mxc), f1 = __expf(ch[25] - mxc);
      const float f2 = __expf(ch[26] - mxc), f3 = __expf(ch[27] - mxc);
      const float fs = f0 + f1 + f2 + f3;
      epi[p * 8 + 0] = f0 / fs; epi[p * 8 + 1] = f1 / fs;
      epi[p * 8 + 2] = f2 / fs; epi[p * 8 + 3] = f3 / fs;
#pragma unroll
      for (int j = 0; j < 4; ++j) epi[p * 8 + 4 + j] = ch[28 + j];
#pragma unroll
      for (int j = 0; j < 4; ++j) actN[p * 32 + j] = (_Float16)ta2[j];
#pragma unroll
      for (int j = 4; j < 16; ++j) actN[p * 32 + j] = (_Float16)0.f;
    }
    ldsFence();

    b = loadB(actN, lane);
    d = wmma16(a04, b, c04);
    leaky8(d);
    storeToB(actN, d, lane);
    ldsFence();

    b = loadB(actN, lane);
    d = wmma16(a05, b, c05);
#pragma unroll
    for (int v = 0; v < 8; ++v) {
      const int m = v + 8 * hi;
      if (m < 4) actP[N * 32 + m] = d[v];
    }
    ldsFence();

    if (lane < 16) {
      const int p = lane;
      float s = 0.f;
#pragma unroll
      for (int c = 0; c < 4; ++c)
        s += actP[p * 32 + c] * epi[p * 8 + c] + epi[p * 8 + 4 + c];
      s = fminf(1.f, fmaxf(-1.f, s));
      const size_t off = (size_t)img * HW + (size_t)row * IMG_W + wt + p;
      const float xv = src[off];
      dst[off] = fmodf(xv - s + 1.0f, 2.0f) - 1.0f;
    }
    ldsFence();
  }
}

// ---- stats: per-channel 256-bin histogram + sum of squares -------------

__global__ void __launch_bounds__(256)
hist_rmse_kernel(const float *__restrict__ src, unsigned *__restrict__ hist,
                 float *__restrict__ sumsq) {
  __shared__ unsigned lh[256];
  __shared__ float red[256];
  const int tid = threadIdx.x;
  lh[tid] = 0;
  __syncthreads();
  const int chan = blockIdx.y;
  const size_t base = (size_t)chan * HW + (size_t)blockIdx.x * 4096;
  float ss = 0.f;
#pragma unroll 4
  for (int i = 0; i < 16; ++i) {
    const float v = src[base + tid + i * 256];
    ss += v * v;
    if (v >= -1.f && v <= 1.f) {
      int idx = (int)floorf((v + 1.f) * 128.f);
      idx = min(255, max(0, idx));
      atomicAdd(&lh[idx], 1u);
    }
  }
  red[tid] = ss;
  __syncthreads();
  for (int s = 128; s > 0; s >>= 1) {
    if (tid < s) red[tid] += red[tid + s];
    __syncthreads();
  }
  if (tid == 0) atomicAdd(sumsq, red[0]);
  atomicAdd(&hist[chan * 256 + tid], lh[tid]);
}

__global__ void __launch_bounds__(256)
finalize_kernel(const unsigned *__restrict__ histX,
                const unsigned *__restrict__ histD,
                const float *__restrict__ sums, float *__restrict__ out) {
  __shared__ float redX[256], redD[256];
  const int tid = threadIdx.x;
  float ex = 0.f, ed = 0.f;
  const float inv = 1.0f / (float)HW;
  for (int i = tid; i < NIMG * 256; i += 256) {
    const float px = (float)histX[i] * inv;
    if (px > 0.f) ex -= px * log2f(px);
    const float pd = (float)histD[i] * inv;
    if (pd > 0.f) ed -= pd * log2f(pd);
  }
  redX[tid] = ex;
  redD[tid] = ed;
  __syncthreads();
  for (int s = 128; s > 0; s >>= 1) {
    if (tid < s) { redX[tid] += redX[tid + s]; redD[tid] += redD[tid + s]; }
    __syncthreads();
  }
  if (tid == 0) {
    const float Nf = (float)((size_t)NIMG * HW);
    out[0] = 128.f * sqrtf(sums[1] / Nf);  // loss1 (residual d)
    out[1] = 128.f * sqrtf(sums[0] / Nf);  // loss0 (x)
    out[2] = redX[0] / (8.f * NIMG);       // inv_cr0
    out[3] = redD[0] / (8.f * NIMG);       // inv_cr1
  }
}

// ------------------------------------------------------------------------

static CParams mkParams(void *const *a) {
  CParams p;
  p.w00T = (const float *)a[0];  p.b00T = (const float *)a[1];
  p.w00L = (const float *)a[2];
  p.w01 = (const float *)a[3];   p.b01 = (const float *)a[4];
  p.w02 = (const float *)a[5];   p.b02 = (const float *)a[6];
  p.w03 = (const float *)a[7];   p.b03 = (const float *)a[8];
  p.w04 = (const float *)a[9];   p.b04 = (const float *)a[10];
  p.w05 = (const float *)a[11];  p.b05 = (const float *)a[12];
  return p;
}

extern "C" void kernel_launch(void *const *d_in, const int *in_sizes, int n_in,
                              void *d_out, int out_size, void *d_ws,
                              size_t ws_size, hipStream_t stream) {
  (void)in_sizes; (void)n_in; (void)out_size; (void)ws_size;
  const float *x = (const float *)d_in[0];
  const CParams p1 = mkParams(d_in + 1);
  const CParams p2 = mkParams(d_in + 14);

  char *ws = (char *)d_ws;
  float *d1 = (float *)ws;                                   // 24*512*512 f32
  float *d2 = (float *)(ws + (size_t)HW * NIMG * 4);         // 24*512*512 f32
  char *statbase = ws + (size_t)HW * NIMG * 8;
  unsigned *histX = (unsigned *)statbase;                    // 24*256 u32
  unsigned *histD = (unsigned *)(statbase + NIMG * 256 * 4); // 24*256 u32
  float *sums = (float *)(statbase + NIMG * 256 * 8);        // [0]=x, [1]=d

  (void)hipMemsetAsync(statbase, 0, NIMG * 256 * 8 + 16, stream);

  const dim3 gP(IMG_H / ROWS_PER_BLOCK, NIMG);
  codec_pass<<<gP, 256, LDS_TOTAL, stream>>>(x, d1, p1);
  codec_pass<<<gP, 256, LDS_TOTAL, stream>>>(d1, d2, p2);

  const dim3 gH(HW / 4096, NIMG);
  hist_rmse_kernel<<<gH, 256, 0, stream>>>(x, histX, &sums[0]);
  hist_rmse_kernel<<<gH, 256, 0, stream>>>(d2, histD, &sums[1]);
  finalize_kernel<<<1, 256, 0, stream>>>(histX, histD, sums, (float *)d_out);
}
